// SparseViT_40742059770376
// MI455X (gfx1250) — matrix operand order repaired and checked by reading, not compiled, not run on hardware
//
#include <hip/hip_runtime.h>
#include <hip/hip_bf16.h>
#include <math.h>

#ifndef __has_builtin
#define __has_builtin(x) 0
#endif

// CDNA5 async global->LDS staging (ASYNCcnt path), with safe fallback.
#if defined(__AMDGCN__) && __has_builtin(__builtin_amdgcn_global_load_async_to_lds_b128) && __has_builtin(__builtin_amdgcn_s_wait_asynccnt)
#define USE_ASYNC_LDS 1
#else
#define USE_ASYNC_LDS 0
#endif

// ---------------- types ----------------
typedef __bf16 bf16;
typedef __attribute__((ext_vector_type(16))) __bf16 bvec16;
typedef __attribute__((ext_vector_type(8)))  __bf16 v8bf;
typedef __attribute__((ext_vector_type(8)))  float  fvec8;

#if USE_ASYNC_LDS
// builtin signature is (global int4*, lds int4*, imm offset, imm cpol) where
// int4 is the GCC vector_size(16) flavor (per the compiler diagnostic).
typedef int v4i_g __attribute__((vector_size(16)));
typedef __attribute__((address_space(1))) v4i_g as1_v4i;
typedef __attribute__((address_space(3))) v4i_g as3_v4i;
__device__ __forceinline__ void async_copy_b128(const void* g, void* l)
{
    // AS1 keeps the full 64-bit VA; AS3 takes low 32 bits (= LDS offset per
    // the flat-address aperture rules). Integer round-trip keeps casts legal.
    __builtin_amdgcn_global_load_async_to_lds_b128(
        (as1_v4i*)(unsigned long long)(size_t)g,
        (as3_v4i*)(unsigned int)(size_t)l, 0, 0);
}
#endif

// ---------------- model dims ----------------
#define BSZ   64
#define CCH   3
#define IMGS  224
#define PS    16
#define NPT   196      // patches per image
#define NTOK  197      // tokens (cls + patches)
#define NTOKP 224      // tokens padded to K-step multiple (32)
#define DM    768
#define HH    12
#define LL    12
#define DFFV  3072
#define NCLS  1000
#define DHD   64
#define SCALE 0.125f   // 64^-0.5
#define NEGV  -1000000000.0f
#define LNEPS 1e-6f

#define BN   (BSZ*NTOK)    // 12608
#define BNP  (BSZ*NPT)     // 12544
#define BH   (BSZ*HH)      // 768

// GEMM tile config: 128(M) x 64(N) x 32(K), 8 wave32 per block
#define MT 128
#define NT 64
#define KT 32

__device__ __forceinline__ int imin(int a, int b) { return a < b ? a : b; }

// =====================================================================
// bf16 WMMA GEMM: C[M,N] = act(A[M,K] @ Bt[N,K]^T + bias) + res
//  - B operand is PRE-TRANSPOSED in memory: Bt[n][k], ldbt >= Kpad
//  - ALL tile loads are unconditional aligned b128 (indices clamped, K
//    padded in the buffers; B's K-pad region is zero where it matters)
//  - tiles staged via GLOBAL_LOAD_ASYNC_TO_LDS_B128 when available
//  - each wave: 16(M) x 64(N) strip -> 4 v_wmma_f32_16x16x32_bf16 / K-step
// =====================================================================
__global__ __launch_bounds__(256)
void gemm_bf16_wmma(const bf16* __restrict__ A, int lda, long long sA,
                    const bf16* __restrict__ Bt, int ldbt, long long sB,
                    float* __restrict__ Cf, int ldc, long long sC,
                    bf16* __restrict__ Cb,
                    const float* __restrict__ bias,
                    const float* __restrict__ res,
                    int M, int N, int K, int act)
{
    __shared__ __align__(16) bf16 ldsA[MT * KT];   // [m][k]
    __shared__ __align__(16) bf16 ldsB[NT * KT];   // [n][k]

    const int tid  = threadIdx.x;
    const int lane = tid & 31;
    const int wid  = tid >> 5;
    const int wm   = wid << 4;                     // wave M strip: 0..112

    const long long bz = blockIdx.z;
    const bf16* ga  = A  + (size_t)(bz * sA);
    const bf16* gbt = Bt + (size_t)(bz * sB);

    const int m0 = blockIdx.y * MT;
    const int n0 = blockIdx.x * NT;

    fvec8 acc0 = {}, acc1 = {}, acc2 = {}, acc3 = {};

    // cooperative stage indices (all unconditional vector ops)
    const int arow = tid >> 1;                     // 0..127
    const int akc  = (tid & 1) << 4;               // 0,16
    const int brow = tid >> 2;                     // 0..63
    const int bkc  = (tid & 3) << 3;               // 0,8,16,24

    const int agm = imin(m0 + arow, M - 1);        // clamped: always in-bounds
    const int bgn = imin(n0 + brow, N - 1);

    // fragment indices (documented WMMA 16-bit lane layouts)
    const int ar  = lane & 15;
    const int akg = (lane >> 4) << 3;              // A: K-group 0/8
    const int bc  = lane & 15;
    const int bkb = (lane >> 4) << 3;              // B: K-group 0/8

    for (int kk = 0; kk < K; kk += KT) {
        const bf16* pa = ga  + (size_t)agm * lda  + (kk + akc);
        const bf16* pb = gbt + (size_t)bgn * ldbt + (kk + bkc);
#if USE_ASYNC_LDS
        // ---- async DMA staging: global -> LDS, no VGPR bounce ----
        async_copy_b128(pa,     &ldsA[arow * KT + akc]);
        async_copy_b128(pa + 8, &ldsA[arow * KT + akc + 8]);
        async_copy_b128(pb,     &ldsB[brow * KT + bkc]);
        __builtin_amdgcn_s_wait_asynccnt(0);
#else
        {
            v8bf a0 = *(const v8bf*)pa;
            v8bf a1 = *(const v8bf*)(pa + 8);
            *(v8bf*)&ldsA[arow * KT + akc]     = a0;
            *(v8bf*)&ldsA[arow * KT + akc + 8] = a1;
            *(v8bf*)&ldsB[brow * KT + bkc] = *(const v8bf*)pb;
        }
#endif
        __syncthreads();

        // prefetch next K-tile (global_prefetch_b8), branch is uniform
        if (kk + KT < K) {
            __builtin_prefetch(ga + (size_t)agm * lda + (kk + KT + akc), 0, 1);
            __builtin_prefetch(gbt + (size_t)bgn * ldbt + (kk + KT + bkc), 0, 1);
        }

        // ---- A fragment: one per wave, reused by 4 WMMAs ----
        const bf16* la = &ldsA[(wm + ar) * KT + akg];
        v8bf alo = *(const v8bf*)la;
        v8bf ahi = *(const v8bf*)(la + 16);
        bvec16 af;
        #pragma unroll
        for (int i = 0; i < 8; ++i) { af[i] = alo[i]; af[i + 8] = ahi[i]; }

        // ---- 4 B fragments + 4 WMMAs ----
        #pragma unroll
        for (int j = 0; j < 4; ++j) {
            const bf16* lb = &ldsB[(j * 16 + bc) * KT + bkb];
            v8bf blo = *(const v8bf*)lb;
            v8bf bhi = *(const v8bf*)(lb + 16);
            bvec16 bfr;
            #pragma unroll
            for (int i = 0; i < 8; ++i) { bfr[i] = blo[i]; bfr[i + 8] = bhi[i]; }
            if      (j == 0) acc0 = __builtin_amdgcn_wmma_f32_16x16x32_bf16(false, af, false, bfr, (short)0, acc0, false, false);
            else if (j == 1) acc1 = __builtin_amdgcn_wmma_f32_16x16x32_bf16(false, af, false, bfr, (short)0, acc1, false, false);
            else if (j == 2) acc2 = __builtin_amdgcn_wmma_f32_16x16x32_bf16(false, af, false, bfr, (short)0, acc2, false, false);
            else             acc3 = __builtin_amdgcn_wmma_f32_16x16x32_bf16(false, af, false, bfr, (short)0, acc3, false, false);
        }
        __syncthreads();
    }

    // ---- epilogue: C/D layout: VGPR r -> row r (lanes 0-15) / r+8 (lanes 16-31)
    const int cc = lane & 15;
    const int rb = (lane >> 4) << 3;
    float*       cf = Cf ? Cf + (size_t)(bz * sC) : nullptr;
    const float* rr = res ? res + (size_t)(bz * sC) : nullptr;
    bf16*        cb = Cb ? Cb + (size_t)(bz * sC) : nullptr;

    #pragma unroll
    for (int j = 0; j < 4; ++j) {
        fvec8 a = (j == 0) ? acc0 : (j == 1) ? acc1 : (j == 2) ? acc2 : acc3;
        const int gn = n0 + j * 16 + cc;
        if (gn < N) {
            const float bvl = bias ? bias[gn] : 0.0f;
            #pragma unroll
            for (int r = 0; r < 8; ++r) {
                const int gm = m0 + wm + rb + r;
                if (gm < M) {
                    float v = a[r] + bvl;
                    if (act) v = 0.5f * v * (1.0f + erff(v * 0.70710678118f));  // exact GELU
                    const size_t idx = (size_t)gm * ldc + gn;
                    if (rr) v += rr[idx];
                    if (cf) cf[idx] = v;
                    if (cb) cb[idx] = (bf16)v;
                }
            }
        }
    }
}

// =====================================================================
// f32 -> bf16 conversion (grid-stride)
// =====================================================================
__global__ void cvt_f32_bf16(const float* __restrict__ s, bf16* __restrict__ d, size_t n)
{
    for (size_t i = (size_t)blockIdx.x * blockDim.x + threadIdx.x; i < n;
         i += (size_t)gridDim.x * blockDim.x)
        d[i] = (bf16)s[i];
}

// batched transpose convert: dst[l][c][r] = src[l][r][c]
__global__ void cvt_transpose_bf16(const float* __restrict__ s, bf16* __restrict__ d,
                                   int Lb, int R, int C)
{
    const size_t total = (size_t)Lb * R * C;
    for (size_t i = (size_t)blockIdx.x * blockDim.x + threadIdx.x; i < total;
         i += (size_t)gridDim.x * blockDim.x) {
        const int l = (int)(i / ((size_t)R * C));
        const int rem = (int)(i % ((size_t)R * C));
        const int r = rem / C, c = rem % C;
        d[(size_t)l * R * C + (size_t)c * R + r] = (bf16)s[i];
    }
}

// zero-fill bf16 buffer
__global__ void zero_bf16(bf16* __restrict__ d, size_t n)
{
    for (size_t i = (size_t)blockIdx.x * blockDim.x + threadIdx.x; i < n;
         i += (size_t)gridDim.x * blockDim.x)
        d[i] = (bf16)0.0f;
}

// =====================================================================
// patchify: x[B,3,224,224] -> bf16 A[B*196, 768] (c*256 + py*16 + px)
// =====================================================================
__global__ void patchify_kernel(const float* __restrict__ x, bf16* __restrict__ out)
{
    const size_t total = (size_t)BNP * DM;
    for (size_t i = (size_t)blockIdx.x * blockDim.x + threadIdx.x; i < total;
         i += (size_t)gridDim.x * blockDim.x) {
        const int m  = (int)(i / DM);
        const int kc = (int)(i % DM);
        const int b  = m / NPT, p = m % NPT;
        const int pi = p / 14, pj = p % 14;
        const int c  = kc / 256, rem = kc % 256;
        const int py = rem / 16, px = rem % 16;
        const size_t src = (((size_t)b * CCH + c) * IMGS + (pi * PS + py)) * IMGS + (pj * PS + px);
        out[i] = (bf16)x[src];
    }
}

// =====================================================================
// token build: tok[b,0] = cls + pos[0]; tok[b,1+p] = patches_f32 + pos[1+p]
// =====================================================================
__global__ void tok_build_kernel(const float* __restrict__ pf, const float* __restrict__ cls,
                                 const float* __restrict__ pos, float* __restrict__ tok)
{
    const size_t total = (size_t)BN * DM;
    for (size_t i = (size_t)blockIdx.x * blockDim.x + threadIdx.x; i < total;
         i += (size_t)gridDim.x * blockDim.x) {
        const int b = (int)(i / ((size_t)NTOK * DM));
        const int rem = (int)(i % ((size_t)NTOK * DM));
        const int n = rem / DM, d = rem % DM;
        float v = (n == 0) ? cls[d] : pf[((size_t)b * NPT + (n - 1)) * DM + d];
        tok[i] = v + pos[(size_t)n * DM + d];
    }
}

// =====================================================================
// token-selection mask -> additive attention bias row per (b, token)
// =====================================================================
__global__ __launch_bounds__(256)
void select_bias_kernel(const float* __restrict__ tok, const float* __restrict__ sel_w,
                        const float* __restrict__ sel_b, float* __restrict__ biasb)
{
    __shared__ float smean[DM];
    __shared__ float sred[8];
    const int b   = blockIdx.x;
    const int tid = threadIdx.x;
    const float* tb = tok + (size_t)b * NTOK * DM;

    for (int d = tid; d < DM; d += 256) {
        float s = 0.f;
        for (int j = 0; j < NPT; ++j) s += tb[(size_t)(1 + j) * DM + d];
        smean[d] = s * (1.0f / (float)NPT);
    }
    __syncthreads();

    float p = 0.f;
    for (int d = tid; d < DM; d += 256) p += smean[d] * sel_w[DM + d];
    #pragma unroll
    for (int o = 16; o > 0; o >>= 1) p += __shfl_xor(p, o, 32);
    if ((tid & 31) == 0) sred[tid >> 5] = p;
    __syncthreads();
    if (tid == 0) {
        float a = 0.f;
        for (int w = 0; w < 8; ++w) a += sred[w];
        sred[0] = a;
    }
    __syncthreads();
    const float meandot = sred[0];

    for (int j = tid; j < NPT; j += 256) {
        float dot = 0.f;
        const float* row = tb + (size_t)(1 + j) * DM;
        for (int d = 0; d < DM; ++d) dot += row[d] * sel_w[d];
        const float logit = dot + meandot + sel_b[0];
        biasb[(size_t)b * NTOK + 1 + j] = (logit > 0.0f) ? 0.0f : NEGV;   // sigmoid(l)>0.5 <=> l>0
    }
    if (tid == 0) biasb[(size_t)b * NTOK] = 0.0f;
}

// =====================================================================
// LayerNorm row kernel: fp32 in -> bf16 out (one block per row, D=768)
// =====================================================================
__global__ __launch_bounds__(256)
void layernorm_kernel(const float* __restrict__ x, const float* __restrict__ g,
                      const float* __restrict__ bta, bf16* __restrict__ out, int D)
{
    __shared__ float red[16];
    const size_t row = blockIdx.x;
    const int tid = threadIdx.x;
    const float* xr = x + row * D;

    float s1 = 0.f, s2 = 0.f;
    for (int i = tid; i < D; i += 256) { float v = xr[i]; s1 += v; s2 += v * v; }
    #pragma unroll
    for (int o = 16; o > 0; o >>= 1) { s1 += __shfl_xor(s1, o, 32); s2 += __shfl_xor(s2, o, 32); }
    if ((tid & 31) == 0) { red[tid >> 5] = s1; red[8 + (tid >> 5)] = s2; }
    __syncthreads();
    if (tid == 0) {
        float a = 0.f, c = 0.f;
        for (int w = 0; w < 8; ++w) { a += red[w]; c += red[8 + w]; }
        red[0] = a; red[8] = c;
    }
    __syncthreads();
    const float mean = red[0] / D;
    const float var  = red[8] / D - mean * mean;
    const float rstd = rsqrtf(var + LNEPS);
    for (int i = tid; i < D; i += 256)
        out[row * D + i] = (bf16)((xr[i] - mean) * rstd * g[i] + bta[i]);
}

// =====================================================================
// qkv split: [B,N,3,H,DH] fp32 ->
//   q [BH][NTOKP][DHD] bf16, k [BH][NTOKP][DHD] bf16 (Bt layout for QK^T),
//   vT[BH][DHD][NTOKP] bf16 (Bt layout for AV; K-pad cols pre-zeroed)
// =====================================================================
__global__ void qkv_split_kernel(const float* __restrict__ qkv, bf16* __restrict__ q,
                                 bf16* __restrict__ k, bf16* __restrict__ vT)
{
    const size_t total = (size_t)BN * DM;
    for (size_t i = (size_t)blockIdx.x * blockDim.x + threadIdx.x; i < total;
         i += (size_t)gridDim.x * blockDim.x) {
        const int b   = (int)(i / ((size_t)NTOK * DM));
        const int rem = (int)(i % ((size_t)NTOK * DM));
        const int n   = rem / DM, hd = rem % DM;
        const int h = hd / DHD, dh = hd % DHD;
        const float* src = qkv + ((size_t)b * NTOK + n) * (3 * DM);
        const size_t qi = (((size_t)(b * HH + h) * NTOKP) + n) * DHD + dh;
        const size_t vi = (((size_t)(b * HH + h) * DHD) + dh) * NTOKP + n;
        q [qi] = (bf16)src[hd];
        k [qi] = (bf16)src[DM + hd];
        vT[vi] = (bf16)src[2 * DM + hd];
    }
}

// =====================================================================
// masked softmax: row = (bh, n); s * SCALE + bias[b, col]; out bf16
// (output row stride NTOKP=224; pad cols are junk but multiply zeroed vT)
// =====================================================================
__global__ __launch_bounds__(256)
void softmax_kernel(const float* __restrict__ s, const float* __restrict__ biasb,
                    bf16* __restrict__ a)
{
    __shared__ float red[8];
    const size_t row = blockIdx.x;                 // BH*NTOK rows
    const int bh = (int)(row / NTOK);
    const int n  = (int)(row % NTOK);
    const int b  = bh / HH;
    const int tid = threadIdx.x;
    const float* sr = s + row * NTOK;

    float v = (tid < NTOK) ? (sr[tid] * SCALE + biasb[(size_t)b * NTOK + tid]) : -INFINITY;

    float mx = v;
    #pragma unroll
    for (int o = 16; o > 0; o >>= 1) mx = fmaxf(mx, __shfl_xor(mx, o, 32));
    if ((tid & 31) == 0) red[tid >> 5] = mx;
    __syncthreads();
    if (tid == 0) {
        float m = red[0];
        for (int w = 1; w < 8; ++w) m = fmaxf(m, red[w]);
        red[0] = m;
    }
    __syncthreads();
    mx = red[0];
    __syncthreads();

    float e = (tid < NTOK) ? __expf(v - mx) : 0.0f;
    float sm = e;
    #pragma unroll
    for (int o = 16; o > 0; o >>= 1) sm += __shfl_xor(sm, o, 32);
    if ((tid & 31) == 0) red[tid >> 5] = sm;
    __syncthreads();
    if (tid == 0) {
        float t = 0.f;
        for (int w = 0; w < 8; ++w) t += red[w];
        red[0] = t;
    }
    __syncthreads();
    const float inv = 1.0f / red[0];
    if (tid < NTOK)
        a[((size_t)bh * NTOK + n) * NTOKP + tid] = (bf16)(e * inv);
}

// =====================================================================
// merge heads: o[BH,NTOK,DH] fp32 -> bf16 [B,N,D]
// =====================================================================
__global__ void merge_heads_kernel(const float* __restrict__ o, bf16* __restrict__ out)
{
    const size_t total = (size_t)BN * DM;
    for (size_t i = (size_t)blockIdx.x * blockDim.x + threadIdx.x; i < total;
         i += (size_t)gridDim.x * blockDim.x) {
        const int b   = (int)(i / ((size_t)NTOK * DM));
        const int rem = (int)(i % ((size_t)NTOK * DM));
        const int n   = rem / DM, hd = rem % DM;
        const int h = hd / DHD, dh = hd % DHD;
        out[i] = (bf16)o[(((size_t)(b * HH + h) * NTOK) + n) * DHD + dh];
    }
}

// =====================================================================
// host-side orchestration
// =====================================================================
extern "C" void kernel_launch(void* const* d_in, const int* in_sizes, int n_in,
                              void* d_out, int out_size, void* d_ws, size_t ws_size,
                              hipStream_t stream)
{
    (void)in_sizes; (void)n_in; (void)out_size; (void)ws_size;
    const float* x        = (const float*)d_in[0];
    const float* patch_w  = (const float*)d_in[1];
    const float* patch_b  = (const float*)d_in[2];
    const float* cls_tok  = (const float*)d_in[3];
    const float* pos_emb  = (const float*)d_in[4];
    const float* sel_w    = (const float*)d_in[5];
    const float* sel_b    = (const float*)d_in[6];
    const float* ln1_g    = (const float*)d_in[7];
    const float* ln1_b    = (const float*)d_in[8];
    const float* qkv_w    = (const float*)d_in[9];
    const float* qkv_b    = (const float*)d_in[10];
    const float* proj_w   = (const float*)d_in[11];
    const float* proj_b   = (const float*)d_in[12];
    const float* ln2_g    = (const float*)d_in[13];
    const float* ln2_b    = (const float*)d_in[14];
    const float* fc1_w    = (const float*)d_in[15];
    const float* fc1_b    = (const float*)d_in[16];
    const float* fc2_w    = (const float*)d_in[17];
    const float* fc2_b    = (const float*)d_in[18];
    const float* norm_g   = (const float*)d_in[19];
    const float* norm_b   = (const float*)d_in[20];
    const float* head_w   = (const float*)d_in[21];
    const float* head_b   = (const float*)d_in[22];

    // ---- carve workspace ----
    char* base = (char*)d_ws;
    size_t off = 0;
    auto alloc = [&](size_t bytes) -> void* {
        void* p = base + off;
        off += (bytes + 255) & ~(size_t)255;
        return p;
    };
    const size_t QKS = (size_t)BH * NTOKP * DHD;     // q/k/vT elem count
    bf16* wqT  = (bf16*)alloc((size_t)LL * 3 * DM * DM * 2);   // [L][2304][768]
    bf16* wprT = (bf16*)alloc((size_t)LL * DM * DM * 2);       // [L][768][768]
    bf16* wf1T = (bf16*)alloc((size_t)LL * DFFV * DM * 2);     // [L][3072][768]
    bf16* wf2T = (bf16*)alloc((size_t)LL * DM * DFFV * 2);     // [L][768][3072]
    bf16* wpB  = (bf16*)alloc((size_t)DM * DM * 2);            // patch_w as-is = Bt
    bf16* whdT = (bf16*)alloc((size_t)NCLS * DM * 2);          // [1000][768]
    bf16* pat_b= (bf16*)alloc((size_t)BNP * DM * 2);
    float* pat_f=(float*)alloc((size_t)BNP * DM * 4);
    float* tok = (float*)alloc((size_t)BN * DM * 4);
    bf16* hbuf = (bf16*)alloc((size_t)BN * DM * 2);
    float* qkvf= (float*)alloc((size_t)BN * 3 * DM * 4);
    bf16* qb   = (bf16*)alloc(QKS * 2);
    bf16* kb   = (bf16*)alloc(QKS * 2);
    bf16* vTb  = (bf16*)alloc(QKS * 2);
    float* sf  = (float*)alloc((size_t)BH * NTOK * NTOK * 4);
    bf16* ab   = (bf16*)alloc((size_t)BH * NTOK * NTOKP * 2);
    float* of  = (float*)alloc((size_t)BH * NTOK * DHD * 4);
    bf16* ob   = (bf16*)alloc((size_t)BN * DM * 2);
    bf16* mlp  = (bf16*)alloc((size_t)BN * DFFV * 2);
    bf16* hfin = (bf16*)alloc((size_t)BN * DM * 2);
    float* biasb=(float*)alloc((size_t)BSZ * NTOK * 4);

    auto gemm = [&](const bf16* A, int lda, long long sA,
                    const bf16* Bt, int ldbt, long long sB,
                    float* Cf, int ldc, long long sC, bf16* Cb,
                    const float* bias, const float* res,
                    int M, int N, int K, int act, int batch) {
        dim3 g((N + NT - 1) / NT, (M + MT - 1) / MT, batch);
        gemm_bf16_wmma<<<g, 256, 0, stream>>>(A, lda, sA, Bt, ldbt, sB,
                                              Cf, ldc, sC, Cb, bias, res,
                                              M, N, K, act);
    };

    // ---- weight conversion (pre-transposed to Bt[n][k]; bf16 fits L2) ----
    cvt_transpose_bf16<<<8192, 256, 0, stream>>>(qkv_w,  wqT,  LL, DM, 3 * DM);
    cvt_transpose_bf16<<<8192, 256, 0, stream>>>(proj_w, wprT, LL, DM, DM);
    cvt_transpose_bf16<<<8192, 256, 0, stream>>>(fc1_w,  wf1T, LL, DM, DFFV);
    cvt_transpose_bf16<<<8192, 256, 0, stream>>>(fc2_w,  wf2T, LL, DFFV, DM);
    cvt_transpose_bf16<<<8192, 256, 0, stream>>>(head_w, whdT, 1, DM, NCLS);
    cvt_f32_bf16<<<4096, 256, 0, stream>>>(patch_w, wpB, (size_t)DM * DM);  // (patch_w.T)^T

    // zero vT once: its K-pad columns (197..223) must stay zero all layers
    zero_bf16<<<4096, 256, 0, stream>>>(vTb, QKS);

    // ---- patch embed ----
    patchify_kernel<<<8192, 256, 0, stream>>>(x, pat_b);
    gemm(pat_b, DM, 0, wpB, DM, 0, pat_f, DM, 0, nullptr, patch_b, nullptr,
         BNP, DM, DM, 0, 1);
    tok_build_kernel<<<8192, 256, 0, stream>>>(pat_f, cls_tok, pos_emb, tok);

    // ---- token selection mask ----
    select_bias_kernel<<<BSZ, 256, 0, stream>>>(tok, sel_w, sel_b, biasb);

    // ---- transformer layers ----
    for (int l = 0; l < LL; ++l) {
        const float* g1 = ln1_g + (size_t)l * DM;
        const float* b1 = ln1_b + (size_t)l * DM;
        const float* g2 = ln2_g + (size_t)l * DM;
        const float* b2 = ln2_b + (size_t)l * DM;
        const bf16*  wq_l  = wqT  + (size_t)l * 3 * DM * DM;
        const float* qb_l  = qkv_b + (size_t)l * 3 * DM;
        const bf16*  wpr_l = wprT + (size_t)l * DM * DM;
        const float* pb_l  = proj_b + (size_t)l * DM;
        const bf16*  wf1_l = wf1T + (size_t)l * DFFV * DM;
        const float* f1b_l = fc1_b + (size_t)l * DFFV;
        const bf16*  wf2_l = wf2T + (size_t)l * DM * DFFV;
        const float* f2b_l = fc2_b + (size_t)l * DM;

        // LN1 -> bf16
        layernorm_kernel<<<BN, 256, 0, stream>>>(tok, g1, b1, hbuf, DM);
        // QKV GEMM
        gemm(hbuf, DM, 0, wq_l, DM, 0, qkvf, 3 * DM, 0, nullptr, qb_l, nullptr,
             BN, 3 * DM, DM, 0, 1);
        qkv_split_kernel<<<8192, 256, 0, stream>>>(qkvf, qb, kb, vTb);
        // S = Q @ K^T : Bt = k (already [n][k] layout), batched over B*H
        gemm(qb, DHD, (long long)NTOKP * DHD,
             kb, DHD, (long long)NTOKP * DHD,
             sf, NTOK, (long long)NTOK * NTOK, nullptr, nullptr, nullptr,
             NTOK, NTOK, DHD, 0, BH);
        // masked softmax -> bf16 A (row stride NTOKP)
        softmax_kernel<<<BH * NTOK, 256, 0, stream>>>(sf, biasb, ab);
        // O = A @ V : Bt = vT [DH][NTOKP], K runs over padded 197->224 (zeros)
        gemm(ab, NTOKP, (long long)NTOK * NTOKP,
             vTb, NTOKP, (long long)DHD * NTOKP,
             of, DHD, (long long)NTOK * DHD, nullptr, nullptr, nullptr,
             NTOK, DHD, NTOK, 0, BH);
        merge_heads_kernel<<<8192, 256, 0, stream>>>(of, ob);
        // proj + residual into tok
        gemm(ob, DM, 0, wpr_l, DM, 0, tok, DM, 0, nullptr, pb_l, tok,
             BN, DM, DM, 0, 1);
        // LN2 -> bf16
        layernorm_kernel<<<BN, 256, 0, stream>>>(tok, g2, b2, hbuf, DM);
        // FC1 + GELU -> bf16
        gemm(hbuf, DM, 0, wf1_l, DM, 0, nullptr, DFFV, 0, mlp, f1b_l, nullptr,
             BN, DFFV, DM, 1, 1);
        // FC2 + residual into tok
        gemm(mlp, DFFV, 0, wf2_l, DFFV, 0, tok, DM, 0, nullptr, f2b_l, tok,
             BN, DM, DFFV, 0, 1);
    }

    // ---- final LN + head (cls token rows only: row stride = NTOK*DM) ----
    layernorm_kernel<<<BN, 256, 0, stream>>>(tok, norm_g, norm_b, hfin, DM);
    gemm(hfin, NTOK * DM, 0, whdT, DM, 0, (float*)d_out, NCLS, 0, nullptr,
         head_b, nullptr, BSZ, NCLS, DM, 0, 1);
}
